// NetVLAD_49306224558124
// MI455X (gfx1250) — matrix operand order repaired and checked by reading, not compiled, not run on hardware
//
#include <hip/hip_runtime.h>

typedef __attribute__((ext_vector_type(16))) _Float16 v16h;
typedef __attribute__((ext_vector_type(8)))  _Float16 v8h;
typedef __attribute__((ext_vector_type(4)))  _Float16 v4h;
typedef __attribute__((ext_vector_type(8)))  float    v8f;

#define ALPHA_F 100.0f
#define Nn 32
#define Cn 128
#define Kk 64
#define Ss 16384
#define TS 64      // pixels per tile
#define TILES 16   // tiles per block -> 1024 px per block
#define SB 16      // s-blocks per batch
#define SCS 136    // stride (f16) for [pixel][c] and centroid [k][c] rows
#define CSS 72     // stride (f16) for [c][pixel] and a[k][pixel] rows

union FragU { v16h v; v8h h[2]; };

// A-matrix fragment (16x32 f16, M x Kdim), row-major storage with given row base.
// lane: m = lane&15 selects row (caller bakes into rowbase), kh = lane>>4.
// VGPR0-3 hold K = k0+8*kh+{0..7}; VGPR4-7 hold K = k0+16+8*kh+{0..7}.
__device__ __forceinline__ v16h loadA(const _Float16* rowbase, int k0, int kh) {
  FragU u;
  u.h[0] = *(const v8h*)(rowbase + k0 + 8 * kh);
  u.h[1] = *(const v8h*)(rowbase + k0 + 16 + 8 * kh);
  return u.v;
}

// B-matrix fragment (32x16 f16, Kdim x N), column-major storage (row = column n).
// lane: column = lane&15 (baked into rowbase), kh = lane>>4 selects K half:
// VGPR0-3 hold K = k0+8*kh+{0..7}; VGPR4-7 hold K = k0+16+8*kh+{0..7}.
__device__ __forceinline__ v16h loadB(const _Float16* rowbase, int k0, int kh) {
  FragU u;
  const _Float16* p = rowbase + k0 + 8 * kh;
  u.h[0] = *(const v8h*)(p);
  u.h[1] = *(const v8h*)(p + 16);
  return u.v;
}

__device__ __forceinline__ v8f wmma_f16(v16h a, v16h b, v8f c) {
  return __builtin_amdgcn_wmma_f32_16x16x32_f16(false, a, false, b, (short)0, c,
                                                false, false);
}

__global__ __launch_bounds__(128)
__attribute__((amdgpu_waves_per_eu(1))) void netvlad_main(
    const float* __restrict__ x, const float* __restrict__ cent,
    float* __restrict__ wv, float* __restrict__ wa) {
  __shared__ _Float16 lds_sc[TS * SCS];   // xn as [pixel][c]
  __shared__ _Float16 lds_cs[Cn * CSS];   // xn as [c][pixel]
  __shared__ _Float16 lds_ct[Kk * SCS];   // 2*alpha*centroids f16 [k][c]
  __shared__ _Float16 lds_a [Kk * CSS];   // softmax probs [k][pixel]
  __shared__ __align__(32) float lds_b[Kk];   // -alpha*||c_k||
  __shared__ float lds_ps[TS * 8];        // per-pixel partial sumsq (8 cgroups)

  const int t    = threadIdx.x;
  const int lane = t & 31;
  const int w    = t >> 5;        // wave id 0..3
  const int mrow = lane & 15;
  const int kh   = lane >> 4;
  const int n    = blockIdx.y;
  const int sb   = blockIdx.x;

  // ---- stage 2*alpha*centroids (f16) + bias b[k] = -alpha*||c_k|| ----
  {
    int k = t >> 1, h = t & 1;
    const float* cr = cent + k * Cn + h * 64;
    float ss = 0.f;
#pragma unroll 8
    for (int i = 0; i < 64; ++i) {
      float v = cr[i];
      ss += v * v;
      lds_ct[k * SCS + h * 64 + i] = (_Float16)(2.f * ALPHA_F * v);
    }
    lds_ps[k * 2 + h] = ss;
  }
  __syncthreads();
  if (t < Kk) lds_b[t] = -ALPHA_F * sqrtf(lds_ps[t * 2] + lds_ps[t * 2 + 1]);

  // persistent VLAD accumulators: wave w owns cluster tile Mt=w, all 8 c-tiles
  v8f acc2[8];
  {
    v8f z = {0.f, 0.f, 0.f, 0.f, 0.f, 0.f, 0.f, 0.f};
#pragma unroll
    for (int i = 0; i < 8; ++i) acc2[i] = z;
  }
  float asum = 0.f;

  const int q4 = t & 15;    // pixel quad: pixels 4*q4 .. 4*q4+3
  const int cg = t >> 4;    // channel group 0..7: channels cg*16 .. cg*16+15
  const float* xb = x + (size_t)n * Cn * Ss;

  for (int tile = 0; tile < TILES; ++tile) {
    __syncthreads();   // protect LDS reuse across iterations
    const int s0 = sb * (TS * TILES) + tile * TS;
    const float* xq = xb + (size_t)(cg * 16) * Ss + s0 + 4 * q4;

    // ---- pass 1: per-pixel sum of squares (b128 streaming loads) ----
    float ss0 = 0.f, ss1 = 0.f, ss2 = 0.f, ss3 = 0.f;
#pragma unroll
    for (int i = 0; i < 16; ++i) {
      float4 v = *(const float4*)(xq + (size_t)i * Ss);
      ss0 += v.x * v.x; ss1 += v.y * v.y;
      ss2 += v.z * v.z; ss3 += v.w * v.w;
    }
    lds_ps[(4 * q4 + 0) * 8 + cg] = ss0;
    lds_ps[(4 * q4 + 1) * 8 + cg] = ss1;
    lds_ps[(4 * q4 + 2) * 8 + cg] = ss2;
    lds_ps[(4 * q4 + 3) * 8 + cg] = ss3;
    __syncthreads();
    float scl[4];
#pragma unroll
    for (int j = 0; j < 4; ++j) {
      float s = 0.f;
#pragma unroll
      for (int g = 0; g < 8; ++g) s += lds_ps[(4 * q4 + j) * 8 + g];
      scl[j] = 1.f / fmaxf(sqrtf(s), 1e-12f);
    }

    // ---- pass 2: normalize, stash both orientations; prefetch next tile ----
#pragma unroll
    for (int i = 0; i < 16; ++i) {
      int c = cg * 16 + i;
      float4 v = *(const float4*)(xq + (size_t)i * Ss);
      _Float16 h0 = (_Float16)(v.x * scl[0]);
      _Float16 h1 = (_Float16)(v.y * scl[1]);
      _Float16 h2 = (_Float16)(v.z * scl[2]);
      _Float16 h3 = (_Float16)(v.w * scl[3]);
      v4h pk = {h0, h1, h2, h3};
      *(v4h*)(lds_cs + c * CSS + 4 * q4) = pk;          // packed b64 store
      lds_sc[(4 * q4 + 0) * SCS + c] = h0;
      lds_sc[(4 * q4 + 1) * SCS + c] = h1;
      lds_sc[(4 * q4 + 2) * SCS + c] = h2;
      lds_sc[(4 * q4 + 3) * SCS + c] = h3;
      __builtin_prefetch(xq + (size_t)i * Ss + TS, 0, 1);  // next tile row
    }
    __syncthreads();

    // ---- matmul1: logits = (2a*cent) @ xn ; wave w -> pixel tile nt=w ----
    v8f lg[4];
    {
      v8f z = {0.f, 0.f, 0.f, 0.f, 0.f, 0.f, 0.f, 0.f};
#pragma unroll
      for (int mt = 0; mt < 4; ++mt) lg[mt] = z;
      const _Float16* brow = lds_sc + (w * 16 + mrow) * SCS;
#pragma unroll
      for (int kc = 0; kc < 4; ++kc) {
        // issue all fragment loads first, then run WMMAs back-to-back
        v16h bf = loadB(brow, kc * 32, kh);
        v16h af[4];
#pragma unroll
        for (int mt = 0; mt < 4; ++mt)
          af[mt] = loadA(lds_ct + (mt * 16 + mrow) * SCS, kc * 32, kh);
#pragma unroll
        for (int mt = 0; mt < 4; ++mt) lg[mt] = wmma_f16(af[mt], bf, lg[mt]);
      }
    }

    // ---- softmax over 64 clusters per pixel (32 local + shfl_xor 16) ----
    float vals[32];
    float mx = -1e30f;
#pragma unroll
    for (int mt = 0; mt < 4; ++mt) {
      v8f bb = *(const v8f*)(lds_b + mt * 16 + 8 * kh);
#pragma unroll
      for (int r = 0; r < 8; ++r) {
        float v = lg[mt][r] + bb[r];
        vals[mt * 8 + r] = v;
        mx = fmaxf(mx, v);
      }
    }
    mx = fmaxf(mx, __shfl_xor(mx, 16));
    float se = 0.f;
#pragma unroll
    for (int i = 0; i < 32; ++i) {
      float e = __expf(vals[i] - mx);
      vals[i] = e;
      se += e;
    }
    se += __shfl_xor(se, 16);
    const float inv = 1.f / se;
#pragma unroll
    for (int mt = 0; mt < 4; ++mt)
#pragma unroll
      for (int r = 0; r < 8; ++r)
        lds_a[(mt * 16 + r + 8 * kh) * CSS + w * 16 + mrow] =
            (_Float16)(vals[mt * 8 + r] * inv);
    __syncthreads();

    // ---- asum accumulation (waves 0-1, wave-uniform branch) ----
    if (t < Kk) {
      const _Float16* ar = lds_a + t * CSS;
      float s = 0.f;
#pragma unroll 8
      for (int i = 0; i < TS; ++i) s += (float)ar[i];
      asum += s;
    }

    // ---- matmul2: vlad += a @ xn^T ; wave w -> cluster tile Mt=w ----
    {
      const _Float16* arow = lds_a + (w * 16 + mrow) * CSS;
#pragma unroll
      for (int kc = 0; kc < 2; ++kc) {
        // preload A and all 8 B fragments, then 8 back-to-back WMMAs
        v16h af = loadA(arow, kc * 32, kh);
        v16h bf[8];
#pragma unroll
        for (int i = 0; i < 8; ++i)
          bf[i] = loadB(lds_cs + (i * 16 + mrow) * CSS, kc * 32, kh);
#pragma unroll
        for (int i = 0; i < 8; ++i) acc2[i] = wmma_f16(af, bf[i], acc2[i]);
      }
    }
  }

  // ---- flush partial results ----
  if (t < Kk) atomicAdd(&wa[n * Kk + t], asum);
#pragma unroll
  for (int i = 0; i < 8; ++i)
#pragma unroll
    for (int r = 0; r < 8; ++r) {
      int k = w * 16 + r + 8 * kh;
      int c = i * 16 + mrow;
      atomicAdd(&wv[((size_t)n * Kk + k) * Cn + c], acc2[i][r]);
    }
}

__global__ __launch_bounds__(256) void netvlad_finalize(
    const float* __restrict__ wv, const float* __restrict__ wa,
    const float* __restrict__ cent, float* __restrict__ out) {
  __shared__ float v[Kk * Cn];      // 32 KB
  __shared__ float red[256];
  __shared__ float kscale[Kk];
  const int t = threadIdx.x;
  const int n = blockIdx.x;
  const int k = t >> 2;   // 4 threads per cluster
  const int q = t & 3;    // 32 channels each

  const float a = wa[n * Kk + k];
  float ss = 0.f;
#pragma unroll 8
  for (int j = 0; j < 32; ++j) {
    int c = q * 32 + j;
    float val = wv[((size_t)n * Kk + k) * Cn + c] - a * cent[k * Cn + c];
    v[k * Cn + c] = val;
    ss += val * val;
  }
  red[t] = ss;
  __syncthreads();
  if (q == 0)
    kscale[k] = 1.f / fmaxf(sqrtf(red[t] + red[t + 1] + red[t + 2] + red[t + 3]),
                            1e-12f);
  __syncthreads();
  const float sc = kscale[k];
  float g = 0.f;
#pragma unroll 8
  for (int j = 0; j < 32; ++j) {
    int c = q * 32 + j;
    float val = v[k * Cn + c] * sc;
    v[k * Cn + c] = val;
    g += val * val;
  }
  red[t] = g;
  __syncthreads();
  for (int off = 128; off > 0; off >>= 1) {
    if (t < off) red[t] += red[t + off];
    __syncthreads();
  }
  const float gsc = 1.f / fmaxf(sqrtf(red[0]), 1e-12f);
#pragma unroll 8
  for (int j = 0; j < 32; ++j) {
    int c = q * 32 + j;
    out[(size_t)n * (Kk * Cn) + k * Cn + c] = v[k * Cn + c] * gsc;
  }
}

extern "C" void kernel_launch(void* const* d_in, const int* in_sizes, int n_in,
                              void* d_out, int out_size, void* d_ws,
                              size_t ws_size, hipStream_t stream) {
  const float* x    = (const float*)d_in[0];
  const float* cent = (const float*)d_in[1];
  float* out = (float*)d_out;
  float* wv  = (float*)d_ws;                       // [N][K][C] f32 partial vlad
  float* wa  = wv + (size_t)Nn * Kk * Cn;          // [N][K] f32 partial asum
  const size_t wsbytes = ((size_t)Nn * Kk * Cn + (size_t)Nn * Kk) * sizeof(float);
  hipMemsetAsync(d_ws, 0, wsbytes, stream);

  dim3 g1(SB, Nn);
  netvlad_main<<<g1, 128, 0, stream>>>(x, cent, wv, wa);
  netvlad_finalize<<<Nn, 256, 0, stream>>>(wv, wa, cent, out);
}
